// VanillaRNN_54984171323420
// MI455X (gfx1250) — matrix operand order repaired and checked by reading, not compiled, not run on hardware
//
#include <hip/hip_runtime.h>

typedef __attribute__((ext_vector_type(16))) __bf16 bf16x16;
typedef __attribute__((ext_vector_type(8)))  float        v8f;
typedef __attribute__((ext_vector_type(4)))  unsigned int uint4v;

#define T_LEN 512
#define STEPS 511          // reference iterates t in [0, T-1)
#define H     256
#define BATCH 2048
#define BN    32           // batch columns per workgroup -> 64 WGs
#define HP    264          // padded stride (bf16 elems) for h columns (+4 banks/lane)
#define XP    132          // padded stride (floats) for x chunk
#define TC    128          // timestep chunk staged in LDS

union FragU { uint4v q[2]; bf16x16 v; };

static __device__ __forceinline__ unsigned int pk2_bf16(float a, float b) {
  unsigned ua = __float_as_uint(a); ua += 0x7FFFu + ((ua >> 16) & 1u);   // RNE
  unsigned ub = __float_as_uint(b); ub += 0x7FFFu + ((ub >> 16) & 1u);
  return (ua >> 16) | (ub & 0xFFFF0000u);
}

static __device__ __forceinline__ float bf2f(unsigned short u) {
  return __uint_as_float(((unsigned int)u) << 16);
}

static __device__ __forceinline__ float fast_tanh(float x) {
  x = fminf(fmaxf(x, -9.0f), 9.0f);
#if __has_builtin(__builtin_amdgcn_tanhf)
  return __builtin_amdgcn_tanhf(x);          // gfx1250 V_TANH_F32
#else
  float e = __expf(2.0f * x);
  return (e - 1.0f) * __builtin_amdgcn_rcpf(e + 1.0f);
#endif
}

static __device__ __forceinline__ v8f wmma_bf16(bf16x16 a, bf16x16 b, v8f c) {
  return __builtin_amdgcn_wmma_f32_16x16x32_bf16(
      /*neg_a=*/false, a, /*neg_b=*/false, b,
      /*c_mod=*/(short)0, c, /*reuse_a=*/false, /*reuse_b=*/false);
}

__global__ __launch_bounds__(256, 1)
void VanillaRNN_wmma_kernel(const float* __restrict__ x,
                            const float* __restrict__ Whx,
                            const float* __restrict__ Whh,
                            const float* __restrict__ Wph,
                            const float* __restrict__ bh,
                            const float* __restrict__ bp,
                            float* __restrict__ out) {
  // ~33 KB + ~16.5 KB of the 320 KB WGP LDS (W_hh lives in VGPRs)
  __shared__ __align__(16) unsigned short s_h[2][BN * HP];    // bf16, column-major, ping-pong
  __shared__ __align__(16) float          s_x[BN * XP];       // x chunk, [col][t]

  const int tid  = threadIdx.x;
  const int lane = tid & 31;
  const int w    = tid >> 5;          // wave id 0..7
  const int l15  = lane & 15;
  const int hi16 = lane >> 4;         // 0 for lanes 0-15, 1 for lanes 16-31
  const int colbase = blockIdx.x * BN;

  const int koffA = hi16 << 3;        // A frag: high lanes hold K+8 within each 16-chunk
  const int koffB = hi16 << 4;        // unused: B frags use koffA, high lanes hold K+8

  // ---- h0 = 0 ----
  for (int i = tid; i < (BN * HP) / 2; i += 256)
    ((unsigned int*)s_h[0])[i] = 0u;

  // ---- W_hh A-fragments: resident in VGPRs for the whole recurrence ----
  // A layout (16-bit 16x32): lane<16 row M=l15 holds K kb..kb+7 (elems 0-7) and
  // kb+16..kb+23 (elems 8-15); lanes>=16 same rows, K offset +8.
  FragU Afrag[2][8];
#pragma unroll
  for (int mi = 0; mi < 2; ++mi) {
    const float* wr = Whh + (32 * w + 16 * mi + l15) * H;
#pragma unroll
    for (int k = 0; k < 8; ++k) {
      const int kb = 32 * k + koffA;
      uint4v lo, hi;
      lo.x = pk2_bf16(wr[kb + 0],  wr[kb + 1]);
      lo.y = pk2_bf16(wr[kb + 2],  wr[kb + 3]);
      lo.z = pk2_bf16(wr[kb + 4],  wr[kb + 5]);
      lo.w = pk2_bf16(wr[kb + 6],  wr[kb + 7]);
      hi.x = pk2_bf16(wr[kb + 16], wr[kb + 17]);
      hi.y = pk2_bf16(wr[kb + 18], wr[kb + 19]);
      hi.z = pk2_bf16(wr[kb + 20], wr[kb + 21]);
      hi.w = pk2_bf16(wr[kb + 22], wr[kb + 23]);
      Afrag[mi][k].q[0] = lo;
      Afrag[mi][k].q[1] = hi;
    }
  }

  // ---- per-lane constant fragments (C-layout: m = 32w + 16*mi + 8*hi16 + r) ----
  float whxf[2][8];
  float bhf[2][2][8];
#pragma unroll
  for (int mi = 0; mi < 2; ++mi) {
    const int mb = 32 * w + 16 * mi + 8 * hi16;
#pragma unroll
    for (int r = 0; r < 8; ++r) {
      whxf[mi][r] = Whx[mb + r];
      for (int nj = 0; nj < 2; ++nj)
        bhf[mi][nj][r] = bh[(mb + r) * BATCH + colbase + 16 * nj + l15];
    }
  }

  const int bA0 = l15 * HP + koffA;          // B frag base offsets (bf16 elems)
  const int bA1 = (16 + l15) * HP + koffA;

  for (int t0 = 0; t0 < STEPS; t0 += TC) {
    __syncthreads();                  // previous chunk's x reads all done
    for (int i = tid; i < BN * TC; i += 256) {
      const int col = i >> 7;         // /TC
      const int tt  = i & (TC - 1);
      s_x[col * XP + tt] = x[(colbase + col) * T_LEN + t0 + tt];
    }
    __syncthreads();

    const int nsteps = (STEPS - t0 < TC) ? (STEPS - t0) : TC;
    for (int s = 0; s < nsteps; ++s) {
      const int t   = t0 + s;
      const int cur = t & 1;
      const unsigned short* hc = s_h[cur];
      unsigned short*       hn = s_h[cur ^ 1];

      const float xv0 = s_x[l15 * XP + s];
      const float xv1 = s_x[(16 + l15) * XP + s];

      // C seeded with b_h + W_hx * x_t (outer product folded into accumulator)
      v8f acc[2][2];
#pragma unroll
      for (int r = 0; r < 8; ++r) {
        acc[0][0][r] = bhf[0][0][r] + whxf[0][r] * xv0;
        acc[0][1][r] = bhf[0][1][r] + whxf[0][r] * xv1;
        acc[1][0][r] = bhf[1][0][r] + whxf[1][r] * xv0;
        acc[1][1][r] = bhf[1][1][r] + whxf[1][r] * xv1;
      }

      // Software-pipelined K loop: B(k+1) loads issue before WMMAs of k,
      // so the 4 XDL ops cover the LDS load-to-use latency.
      FragU Bd[2][2];                  // [buffer][nj]
      Bd[0][0].q[0] = *(const uint4v*)&hc[bA0];
      Bd[0][0].q[1] = *(const uint4v*)&hc[bA0 + 16];
      Bd[0][1].q[0] = *(const uint4v*)&hc[bA1];
      Bd[0][1].q[1] = *(const uint4v*)&hc[bA1 + 16];
#pragma unroll
      for (int k = 0; k < 8; ++k) {
        const int cb = k & 1, nb = cb ^ 1;
        if (k < 7) {
          const int kb2 = (k + 1) << 5;
          Bd[nb][0].q[0] = *(const uint4v*)&hc[bA0 + kb2];
          Bd[nb][0].q[1] = *(const uint4v*)&hc[bA0 + kb2 + 16];
          Bd[nb][1].q[0] = *(const uint4v*)&hc[bA1 + kb2];
          Bd[nb][1].q[1] = *(const uint4v*)&hc[bA1 + kb2 + 16];
        }
        acc[0][0] = wmma_bf16(Afrag[0][k].v, Bd[cb][0].v, acc[0][0]);
        acc[0][1] = wmma_bf16(Afrag[0][k].v, Bd[cb][1].v, acc[0][1]);
        acc[1][0] = wmma_bf16(Afrag[1][k].v, Bd[cb][0].v, acc[1][0]);
        acc[1][1] = wmma_bf16(Afrag[1][k].v, Bd[cb][1].v, acc[1][1]);
      }

      // tanh + pack + store h_{t+1} (matches C fragment VGPR->(M,N) mapping)
#pragma unroll
      for (int mi = 0; mi < 2; ++mi) {
        const int rowst = 32 * w + 16 * mi + 8 * hi16;
#pragma unroll
        for (int nj = 0; nj < 2; ++nj) {
          const int col = 16 * nj + l15;
          uint4v p;
          p.x = pk2_bf16(fast_tanh(acc[mi][nj][0]), fast_tanh(acc[mi][nj][1]));
          p.y = pk2_bf16(fast_tanh(acc[mi][nj][2]), fast_tanh(acc[mi][nj][3]));
          p.z = pk2_bf16(fast_tanh(acc[mi][nj][4]), fast_tanh(acc[mi][nj][5]));
          p.w = pk2_bf16(fast_tanh(acc[mi][nj][6]), fast_tanh(acc[mi][nj][7]));
          *(uint4v*)&hn[col * HP + rowst] = p;
        }
      }
      __syncthreads();                // one barrier per step (ping-pong buffers)
    }
  }

  // ---- output projection: out[b][c] = W_ph[c,:] . h_final[:,b] + b_p[c][b] ----
  const unsigned short* hf = s_h[STEPS & 1];
  for (int idx = tid; idx < BN * 10; idx += 256) {
    const int col = idx / 10;
    const int c   = idx % 10;
    float a = 0.0f;
    for (int m = 0; m < H; ++m)
      a += Wph[c * H + m] * bf2f(hf[col * HP + m]);
    const int b = colbase + col;
    out[b * 10 + c] = a + bp[c * BATCH + b];
  }
}

extern "C" void kernel_launch(void* const* d_in, const int* in_sizes, int n_in,
                              void* d_out, int out_size, void* d_ws, size_t ws_size,
                              hipStream_t stream) {
  const float* x   = (const float*)d_in[0];
  const float* Whx = (const float*)d_in[1];
  const float* Whh = (const float*)d_in[2];
  const float* Wph = (const float*)d_in[3];
  const float* bh  = (const float*)d_in[4];
  const float* bp  = (const float*)d_in[5];
  float* out = (float*)d_out;
  hipLaunchKernelGGL(VanillaRNN_wmma_kernel, dim3(BATCH / BN), dim3(256), 0, stream,
                     x, Whx, Whh, Wph, bh, bp, out);
}
